// CausalSelfAttention_4664334483648
// MI455X (gfx1250) — matrix operand order repaired and checked by reading, not compiled, not run on hardware
//
#include <hip/hip_runtime.h>

// ---------- types ----------
typedef __attribute__((ext_vector_type(16))) __bf16 v16bf;
typedef __attribute__((ext_vector_type(8)))  __bf16 v8bf;
typedef __attribute__((ext_vector_type(4)))  __bf16 v4bf;
typedef __attribute__((ext_vector_type(8)))  float  v8f;
typedef __attribute__((ext_vector_type(4)))  float  v4f;
typedef __attribute__((ext_vector_type(4)))  int    v4i;

#define DEVI __device__ __forceinline__

#if __has_builtin(__builtin_amdgcn_global_load_async_to_lds_b128)
#define HAVE_ASYNC_LDS 1
typedef __attribute__((address_space(1))) v4i gv4i_t;  // global int4
typedef __attribute__((address_space(3))) v4i lv4i_t;  // LDS int4
#else
#define HAVE_ASYNC_LDS 0
#endif

DEVI v16bf join8(v8bf lo, v8bf hi) {
  return __builtin_shufflevector(lo, hi, 0,1,2,3,4,5,6,7,8,9,10,11,12,13,14,15);
}

DEVI v8f wmma_bf16(v16bf a, v16bf b, v8f c) {
  // D = A(16x32 bf16) * B(32x16 bf16) + C(16x16 f32)
  return __builtin_amdgcn_wmma_f32_16x16x32_bf16(false, a, false, b, (short)0, c,
                                                 false, false);
}

// cross-lane reduction within each 16-lane half of the wave32 (xor 1,2,4,8)
#define SWZF(x, pat) __int_as_float(__builtin_amdgcn_ds_swizzle(__float_as_int(x), (pat)))

#define NEGBIG (-1.0e30f)

// ---------- f32 -> bf16 conversion ----------
__global__ void cvt_f32_bf16(const float* __restrict__ src,
                             __bf16* __restrict__ dst, int n) {
  int i = (blockIdx.x * blockDim.x + threadIdx.x) * 4;
  if (i < n) {
    v4f f = *(const v4f*)(src + i);
    v4bf o;
    o.x = (__bf16)f.x; o.y = (__bf16)f.y; o.z = (__bf16)f.z; o.w = (__bf16)f.w;
    *(v4bf*)(dst + i) = o;
  }
}

// ---------- GEMM: out = A[M,K] @ W[N,K]^T + bias[N] ----------
// mode 0: bf16 out scattered to [B,H,T,D]   (q, k)
// mode 1: bf16 out scattered to [B,H,D,T]   (v, transposed for PV B-operand)
// mode 3: f32  out linear [M,N]             (final projection -> d_out)
__global__ __launch_bounds__(128) void gemm_bf16(
    const __bf16* __restrict__ A, const __bf16* __restrict__ W,
    const float* __restrict__ bias, void* __restrict__ outp,
    int M, int N, int K, int mode) {
  const int lane = threadIdx.x & 31;
  const int wave = threadIdx.x >> 5;
  const int lm = lane & 15, hf = lane >> 4;
  const int Rm = blockIdx.x * 64 + (wave >> 1) * 32;
  const int Cn = blockIdx.y * 128 + (wave & 1) * 64;

  v8f acc[2][4];
  const v8f vzero = {0.f, 0.f, 0.f, 0.f, 0.f, 0.f, 0.f, 0.f};
#pragma unroll
  for (int mi = 0; mi < 2; mi++)
#pragma unroll
    for (int ni = 0; ni < 4; ni++) acc[mi][ni] = vzero;

  for (int kk = 0; kk < K; kk += 32) {
    v16bf af[2], bfr[4];
#pragma unroll
    for (int mi = 0; mi < 2; mi++) {
      const __bf16* ar = A + (size_t)(Rm + mi * 16 + lm) * K + kk + hf * 8;
      af[mi] = join8(*(const v8bf*)ar, *(const v8bf*)(ar + 16));
    }
#pragma unroll
    for (int ni = 0; ni < 4; ni++) {
      const __bf16* br = W + (size_t)(Cn + ni * 16 + lm) * K + kk + hf * 8;
      bfr[ni] = join8(*(const v8bf*)br, *(const v8bf*)(br + 16));
    }
    if (kk + 32 < K) {
      __builtin_prefetch(A + (size_t)(Rm + lm) * K + kk + 32, 0, 1);
      __builtin_prefetch(W + (size_t)(Cn + lm) * K + kk + 32, 0, 1);
    }
#pragma unroll
    for (int mi = 0; mi < 2; mi++)
#pragma unroll
      for (int ni = 0; ni < 4; ni++)
        acc[mi][ni] = wmma_bf16(af[mi], bfr[ni], acc[mi][ni]);
  }

#pragma unroll
  for (int mi = 0; mi < 2; mi++) {
#pragma unroll
    for (int ni = 0; ni < 4; ni++) {
      int n = Cn + ni * 16 + lm;
      float bn = bias[n];
      int hH = n >> 6, dd = n & 63;  // head, dim (C layout: n = h*64+d)
#pragma unroll
      for (int r = 0; r < 8; r++) {
        int m = Rm + mi * 16 + r + 8 * hf;  // C/D layout: M = r + 8*(lane/16)
        float val = acc[mi][ni][r] + bn;
        if (mode == 0) {
          int bb = m >> 11, tt = m & 2047;
          ((__bf16*)outp)[(((size_t)bb * 16 + hH) * 2048 + tt) * 64 + dd] =
              (__bf16)val;
        } else if (mode == 1) {
          int bb = m >> 11, tt = m & 2047;
          ((__bf16*)outp)[(((size_t)bb * 16 + hH) * 64 + dd) * 2048 + tt] =
              (__bf16)val;
        } else {
          ((float*)outp)[(size_t)m * N + n] = val;
        }
      }
    }
  }
}

// ---------- flash attention ----------
// Q,K: [B*H, T, 64] bf16 ; Vt: [B*H, 64, T] bf16 ; Y: [B, T, C] bf16
// K/V tiles for each 64-key block are staged once per workgroup into LDS via
// async global->LDS loads (ASYNCcnt), then all 4 waves feed WMMA from LDS.
#define LDS_STRIDE 72  // 64 + 8 pad: 144B rows -> conflict-free-ish b128 reads

__global__ __launch_bounds__(128) void flash_attn(
    const __bf16* __restrict__ Q, const __bf16* __restrict__ Kc,
    const __bf16* __restrict__ Vt, __bf16* __restrict__ Y) {
  const int T = 2048;
  const int bh = blockIdx.x;
  const int qblk = blockIdx.y;
  const int tid = threadIdx.x;
  const int lane = tid & 31;
  const int wave = tid >> 5;
  const int lm = lane & 15, hf = lane >> 4;
  const int qbase = qblk * 64 + wave * 16;

  const __bf16* qh = Q + (size_t)bh * T * 64;
  const __bf16* kh = Kc + (size_t)bh * T * 64;
  const __bf16* vh = Vt + (size_t)bh * 64 * T;

  __shared__ __align__(16) __bf16 kbuf[64 * LDS_STRIDE];  // [s_local][d]
  __shared__ __align__(16) __bf16 vbuf[64 * LDS_STRIDE];  // [d][s_local]
  __shared__ __align__(16) __bf16 pbuf[4][16 * 64];       // per-wave p tile
  __bf16* pb = pbuf[wave];

  // staging assignment: thread -> (row 0..63, 32-col half)
  const int crow = tid >> 1;
  const int chalf = (tid & 1) * 32;

  // preload q A-fragments (K = head dim 64 -> two 32-chunks)
  v16bf aq[2];
#pragma unroll
  for (int ks = 0; ks < 2; ks++) {
    const __bf16* qr = qh + (size_t)(qbase + lm) * 64 + ks * 32 + hf * 8;
    aq[ks] = join8(*(const v8bf*)qr, *(const v8bf*)(qr + 16));
  }

  const v8f vzero = {0.f, 0.f, 0.f, 0.f, 0.f, 0.f, 0.f, 0.f};
  v8f O[4];
#pragma unroll
  for (int dt = 0; dt < 4; dt++) O[dt] = vzero;
  float rmax[8], rsum[8];
#pragma unroll
  for (int r = 0; r < 8; r++) { rmax[r] = NEGBIG; rsum[r] = 0.f; }

  const float scl = 0.125f * 1.44269504088896340736f;  // 1/sqrt(64) * log2(e)

  for (int sb = 0; sb <= qblk; ++sb) {
    const int sbase = sb * 64;

    // ---- stage K[s,d] and V^T[d,s] tiles into LDS ----
    __syncthreads();  // prior iteration's LDS readers are done
#if HAVE_ASYNC_LDS
#pragma unroll
    for (int j = 0; j < 4; j++) {
      __builtin_amdgcn_global_load_async_to_lds_b128(
          (gv4i_t*)(kh + (size_t)(sbase + crow) * 64 + chalf + j * 8),
          (lv4i_t*)(kbuf + crow * LDS_STRIDE + chalf + j * 8), 0, 0);
      __builtin_amdgcn_global_load_async_to_lds_b128(
          (gv4i_t*)(vh + (size_t)crow * T + sbase + chalf + j * 8),
          (lv4i_t*)(vbuf + crow * LDS_STRIDE + chalf + j * 8), 0, 0);
    }
#if __has_builtin(__builtin_amdgcn_s_wait_asynccnt)
    __builtin_amdgcn_s_wait_asynccnt(0);
#else
    __asm__ volatile("s_wait_asynccnt 0" ::: "memory");
#endif
#else
#pragma unroll
    for (int j = 0; j < 4; j++) {
      *(v8bf*)(kbuf + crow * LDS_STRIDE + chalf + j * 8) =
          *(const v8bf*)(kh + (size_t)(sbase + crow) * 64 + chalf + j * 8);
      *(v8bf*)(vbuf + crow * LDS_STRIDE + chalf + j * 8) =
          *(const v8bf*)(vh + (size_t)crow * T + sbase + chalf + j * 8);
    }
#endif
    __syncthreads();

    // ---- scores: q[16x64] @ k[64x64]^T (B-operand rows from LDS) ----
    v8f sc[4];
#pragma unroll
    for (int nt = 0; nt < 4; nt++) sc[nt] = vzero;
#pragma unroll
    for (int ks = 0; ks < 2; ks++) {
#pragma unroll
      for (int nt = 0; nt < 4; nt++) {
        const __bf16* kr = kbuf + (nt * 16 + lm) * LDS_STRIDE + ks * 32 + hf * 8;
        v16bf bk = join8(*(const v8bf*)kr, *(const v8bf*)(kr + 16));
        sc[nt] = wmma_bf16(aq[ks], bk, sc[nt]);
      }
    }

    // scale (folded log2e) + causal mask
#pragma unroll
    for (int nt = 0; nt < 4; nt++) {
      int s = sbase + nt * 16 + lm;
#pragma unroll
      for (int r = 0; r < 8; r++) {
        int t = qbase + r + 8 * hf;
        float x = sc[nt][r] * scl;
        sc[nt][r] = (s <= t) ? x : NEGBIG;
      }
    }

    // block row-max, reduced across the 16-lane half
    float bm[8];
#pragma unroll
    for (int r = 0; r < 8; r++)
      bm[r] = fmaxf(fmaxf(sc[0][r], sc[1][r]), fmaxf(sc[2][r], sc[3][r]));
#pragma unroll
    for (int r = 0; r < 8; r++) {
      bm[r] = fmaxf(bm[r], SWZF(bm[r], 0x041f));
      bm[r] = fmaxf(bm[r], SWZF(bm[r], 0x081f));
      bm[r] = fmaxf(bm[r], SWZF(bm[r], 0x101f));
      bm[r] = fmaxf(bm[r], SWZF(bm[r], 0x201f));
    }

    float corr[8];
#pragma unroll
    for (int r = 0; r < 8; r++) {
      float nm = fmaxf(rmax[r], bm[r]);
      corr[r] = exp2f(rmax[r] - nm);
      rmax[r] = nm;
    }

    // p = 2^(sc - rowmax), block row-sum
    float bs[8];
#pragma unroll
    for (int r = 0; r < 8; r++) bs[r] = 0.f;
#pragma unroll
    for (int nt = 0; nt < 4; nt++)
#pragma unroll
      for (int r = 0; r < 8; r++) {
        float p = exp2f(sc[nt][r] - rmax[r]);
        sc[nt][r] = p;
        bs[r] += p;
      }
#pragma unroll
    for (int r = 0; r < 8; r++) {
      bs[r] += SWZF(bs[r], 0x041f);
      bs[r] += SWZF(bs[r], 0x081f);
      bs[r] += SWZF(bs[r], 0x101f);
      bs[r] += SWZF(bs[r], 0x201f);
      rsum[r] = rsum[r] * corr[r] + bs[r];
    }

    // rescale running output
#pragma unroll
    for (int dt = 0; dt < 4; dt++)
#pragma unroll
      for (int r = 0; r < 8; r++) O[dt][r] *= corr[r];

    // stage p (C/D layout) into LDS, re-read in A layout
#pragma unroll
    for (int nt = 0; nt < 4; nt++)
#pragma unroll
      for (int r = 0; r < 8; r++)
        pb[(r + 8 * hf) * 64 + nt * 16 + lm] = (__bf16)sc[nt][r];

    __asm__ volatile("s_wait_dscnt 0" ::: "memory");

    // ---- O += p[16x64] @ v[64x64] (B-operand lane n = d, rows of V^T) ----
#pragma unroll
    for (int ks = 0; ks < 2; ks++) {
      const __bf16* pr = pb + lm * 64 + ks * 32 + hf * 8;
      v16bf ap = join8(*(const v8bf*)pr, *(const v8bf*)(pr + 16));
#pragma unroll
      for (int dt = 0; dt < 4; dt++) {
        const __bf16* vr = vbuf + (dt * 16 + lm) * LDS_STRIDE + ks * 32 + hf * 8;
        v16bf bv = join8(*(const v8bf*)vr, *(const v8bf*)(vr + 16));
        O[dt] = wmma_bf16(ap, bv, O[dt]);
      }
    }
  }

  // normalize and write y in [B,T,C] bf16
  const int bb = bh >> 4, hh = bh & 15;
#pragma unroll
  for (int r = 0; r < 8; r++) {
    float inv = 1.0f / rsum[r];
    int t = qbase + r + 8 * hf;
#pragma unroll
    for (int dt = 0; dt < 4; dt++) {
      int col = hh * 64 + dt * 16 + lm;
      Y[((size_t)bb * 2048 + t) * 1024 + col] = (__bf16)(O[dt][r] * inv);
    }
  }
}

// ---------- host ----------
extern "C" void kernel_launch(void* const* d_in, const int* in_sizes, int n_in,
                              void* d_out, int out_size, void* d_ws,
                              size_t ws_size, hipStream_t stream) {
  const float* x  = (const float*)d_in[0];
  const float* Wq = (const float*)d_in[1];
  const float* bq = (const float*)d_in[2];
  const float* Wk = (const float*)d_in[3];
  const float* bk = (const float*)d_in[4];
  const float* Wv = (const float*)d_in[5];
  const float* bv = (const float*)d_in[6];
  const float* Wp = (const float*)d_in[7];
  const float* bp = (const float*)d_in[8];
  float* out = (float*)d_out;

  const int M = 8192, C = 1024;  // M = B*T
  char* ws = (char*)d_ws;
  __bf16* xb  = (__bf16*)ws; ws += (size_t)M * C * 2;
  __bf16* wqb = (__bf16*)ws; ws += (size_t)C * C * 2;
  __bf16* wkb = (__bf16*)ws; ws += (size_t)C * C * 2;
  __bf16* wvb = (__bf16*)ws; ws += (size_t)C * C * 2;
  __bf16* wpb = (__bf16*)ws; ws += (size_t)C * C * 2;
  __bf16* qb  = (__bf16*)ws; ws += (size_t)M * C * 2;  // [B,H,T,D]
  __bf16* kb  = (__bf16*)ws; ws += (size_t)M * C * 2;  // [B,H,T,D]
  __bf16* vtb = (__bf16*)ws; ws += (size_t)M * C * 2;  // [B,H,D,T]
  __bf16* yb  = xb;  // reuse: x_bf16 dead after v projection

  const int nxc = M * C, nwc = C * C;
  cvt_f32_bf16<<<nxc / 1024, 256, 0, stream>>>(x, xb, nxc);
  cvt_f32_bf16<<<nwc / 1024, 256, 0, stream>>>(Wq, wqb, nwc);
  cvt_f32_bf16<<<nwc / 1024, 256, 0, stream>>>(Wk, wkb, nwc);
  cvt_f32_bf16<<<nwc / 1024, 256, 0, stream>>>(Wv, wvb, nwc);
  cvt_f32_bf16<<<nwc / 1024, 256, 0, stream>>>(Wp, wpb, nwc);

  dim3 gblk(M / 64, C / 128);
  gemm_bf16<<<gblk, 128, 0, stream>>>(xb, wqb, bq, qb, M, C, C, 0);
  gemm_bf16<<<gblk, 128, 0, stream>>>(xb, wkb, bk, kb, M, C, C, 0);
  gemm_bf16<<<gblk, 128, 0, stream>>>(xb, wvb, bv, vtb, M, C, C, 1);

  flash_attn<<<dim3(64, 32), 128, 0, stream>>>(qb, kb, vtb, yb);

  gemm_bf16<<<gblk, 128, 0, stream>>>(yb, wpb, bp, out, M, C, C, 3);
}
